// SAGAN_50577534877744
// MI455X (gfx1250) — matrix operand order repaired and checked by reading, not compiled, not run on hardware
//
#include <hip/hip_runtime.h>
#include <hip/hip_bf16.h>

typedef __bf16 bf16;
typedef __bf16 v16bf __attribute__((ext_vector_type(16)));
typedef float  v8f   __attribute__((ext_vector_type(8)));

namespace {
constexpr int cN = 16384, cB = 128, cNPG = 128, cDEG = 8, cF = 128;
constexpr int cE  = cN * cDEG;      // 131072 edges
constexpr int cET = cE + cN;        // + self loops = 147456
constexpr int cD = 544, cK = 30, cND = 2048;

enum { GF_TRANSA = 1, GF_BIAS = 2, GF_RELU = 4 };

struct BfPack { uint4 a, b; };      // 32 bytes == v16bf
}

__device__ inline unsigned sg_packbf(float lo, float hi) {
  bf16 a = (bf16)lo, b = (bf16)hi;
  unsigned short ua = __builtin_bit_cast(unsigned short, a);
  unsigned short ub = __builtin_bit_cast(unsigned short, b);
  return ((unsigned)ub << 16) | ua;
}

// ---------------------------------------------------------------------------
// Batched WMMA GEMM: C[b] = op(A[b]) * B[b] (+bias) (+relu).
// fp32 in/out, bf16 multiply (v_wmma_f32_16x16x32_bf16).
// Block = 64M x 64N macro tile, 4 waves; B panel staged in LDS (bf16,
// transposed, pair-packed) shared by all waves; each wave runs 4 WMMAs per
// 32-K step off one A fragment.  Requires K % 32 == 0.  The K loop is
// branch-free (index clamping); the epilogue has a block-uniform interior
// fast path with unguarded pointer-stepped stores.
// ---------------------------------------------------------------------------
template <int FLAGS>
__global__ __launch_bounds__(128) void sg_wmma_gemm(
    const float* __restrict__ A, const float* __restrict__ Bg,
    const float* __restrict__ bias, float* __restrict__ C,
    int M, int N, int K, int lda, int ldb, int ldc,
    long long sA, long long sB, long long sC)
{
  __shared__ unsigned blds[1024];   // 64 N-rows x 16 dwords (32 bf16 K values)
  const int tid  = threadIdx.x;
  const int wave = tid >> 5;
  const int lane = tid & 31;
  const int tilesN = (N + 63) >> 6;
  const int blkM = (blockIdx.x / tilesN) << 6;
  const int blkN = (blockIdx.x % tilesN) << 6;

  A  += (long long)blockIdx.y * sA;
  Bg += (long long)blockIdx.y * sB;
  C  += (long long)blockIdx.y * sC;

  int am = blkM + (wave << 4) + (lane & 15);
  am = am < M ? am : M - 1;                  // clamp: load-safe, store-masked
  const int khA = (lane >> 4) << 3;          // A: lanes 16-31 hold K+8 group
  const int khB = (lane >> 4) << 3;          // B: lanes 16-31 hold K+8 group

  v8f acc0 = {0.f,0.f,0.f,0.f,0.f,0.f,0.f,0.f};
  v8f acc1 = acc0, acc2 = acc0, acc3 = acc0;

  for (int kb = 0; kb < K; kb += 32) {
    // ---- cooperatively stage B(kb:kb+32, blkN:blkN+64) into LDS ----
#pragma unroll
    for (int j = 0; j < 8; ++j) {
      int p  = tid + (j << 7);               // 0..1023 packed K-pairs
      int n  = p & 63;
      int k2 = (p >> 6) << 1;                // 0,2,...,30
      int nc = blkN + n; nc = nc < N ? nc : N - 1;
      const float* bp = Bg + (long long)(kb + k2) * ldb + nc;
      float v0 = bp[0];
      float v1 = bp[ldb];
      blds[(n << 4) + (k2 >> 1)] = sg_packbf(v0, v1);
    }

    // ---- A fragment (per-wave 16 rows) ----
    v16bf af;
    if (FLAGS & GF_TRANSA) {
#pragma unroll
      for (int i = 0; i < 16; ++i) {
        int kl = ((i & 8) ? 16 : 0) + khA + (((i >> 1) & 3) << 1) + (i & 1);
        af[i] = (bf16)A[(long long)(kb + kl) * lda + am];
      }
    } else {
      const float* ap = A + (long long)am * lda + kb;
      float4 f0 = *(const float4*)(ap + khA);
      float4 f1 = *(const float4*)(ap + khA + 4);
      float4 f2 = *(const float4*)(ap + 16 + khA);
      float4 f3 = *(const float4*)(ap + 16 + khA + 4);
      af[0]=(bf16)f0.x;  af[1]=(bf16)f0.y;  af[2]=(bf16)f0.z;  af[3]=(bf16)f0.w;
      af[4]=(bf16)f1.x;  af[5]=(bf16)f1.y;  af[6]=(bf16)f1.z;  af[7]=(bf16)f1.w;
      af[8]=(bf16)f2.x;  af[9]=(bf16)f2.y;  af[10]=(bf16)f2.z; af[11]=(bf16)f2.w;
      af[12]=(bf16)f3.x; af[13]=(bf16)f3.y; af[14]=(bf16)f3.z; af[15]=(bf16)f3.w;
    }

    __syncthreads();

    // ---- 4 B fragments from LDS, 4 WMMAs ----
    {
      const int nrow = lane & 15;
      const unsigned* base = blds + (khB >> 1);
      const uint4* q0 = (const uint4*)(base + ((nrow      ) << 4));
      const uint4* q1 = (const uint4*)(base + ((nrow + 16 ) << 4));
      const uint4* q2 = (const uint4*)(base + ((nrow + 32 ) << 4));
      const uint4* q3 = (const uint4*)(base + ((nrow + 48 ) << 4));
      BfPack p0; p0.a = q0[0]; p0.b = q0[2];
      BfPack p1; p1.a = q1[0]; p1.b = q1[2];
      BfPack p2; p2.a = q2[0]; p2.b = q2[2];
      BfPack p3; p3.a = q3[0]; p3.b = q3[2];
      v16bf b0 = __builtin_bit_cast(v16bf, p0);
      v16bf b1 = __builtin_bit_cast(v16bf, p1);
      v16bf b2 = __builtin_bit_cast(v16bf, p2);
      v16bf b3 = __builtin_bit_cast(v16bf, p3);
      acc0 = __builtin_amdgcn_wmma_f32_16x16x32_bf16(false, af, false, b0, (short)0, acc0, false, false);
      acc1 = __builtin_amdgcn_wmma_f32_16x16x32_bf16(false, af, false, b1, (short)0, acc1, false, false);
      acc2 = __builtin_amdgcn_wmma_f32_16x16x32_bf16(false, af, false, b2, (short)0, acc2, false, false);
      acc3 = __builtin_amdgcn_wmma_f32_16x16x32_bf16(false, af, false, b3, (short)0, acc3, false, false);
    }
    __syncthreads();
  }

  // ---- epilogue (C layout: vgpr r -> row m0 + (lane>=16? 8:0) + r) ----
  const int cmB = blkM + (wave << 4) + ((lane >> 4) << 3);
  const int cnL = lane & 15;

  if (blkM + 64 <= M && blkN + 64 <= N) {
    // interior fast path: no guards, one base pointer per fragment
#pragma unroll
    for (int f = 0; f < 4; ++f) {
      const v8f& acc = f == 0 ? acc0 : f == 1 ? acc1 : f == 2 ? acc2 : acc3;
      int cn = blkN + (f << 4) + cnL;
      float bv = 0.f;
      if (FLAGS & GF_BIAS) bv = bias[cn];
      float* cp = C + (long long)cmB * ldc + cn;
#pragma unroll
      for (int r = 0; r < 8; ++r) {
        float v = acc[r] + bv;
        if (FLAGS & GF_RELU) v = v > 0.f ? v : 0.f;
        *cp = v;
        cp += ldc;
      }
    }
  } else {
#pragma unroll
    for (int f = 0; f < 4; ++f) {
      const v8f& acc = f == 0 ? acc0 : f == 1 ? acc1 : f == 2 ? acc2 : acc3;
      int cn = blkN + (f << 4) + cnL;
      float bv = 0.f;
      if ((FLAGS & GF_BIAS) && cn < N) bv = bias[cn];
      float* cp = C + (long long)cmB * ldc + cn;
#pragma unroll
      for (int r = 0; r < 8; ++r) {
        if (cmB + r < M && cn < N) {
          float v = acc[r] + bv;
          if (FLAGS & GF_RELU) v = v > 0.f ? v : 0.f;
          *cp = v;
        }
        cp += ldc;
      }
    }
  }
}

// ---------------------------------------------------------------------------
// Elementwise / segment kernels
// ---------------------------------------------------------------------------
__global__ void sg_zero(float* p, long long n) {
  long long i = (long long)blockIdx.x * 256 + threadIdx.x;
  if (i < n) p[i] = 0.f;
}

__device__ inline unsigned sg_flip(float f) {
  unsigned u = __float_as_uint(f);
  return (u >> 31) ? ~u : (u | 0x80000000u);
}
__device__ inline float sg_unflip(unsigned u) {
  return __uint_as_float((u >> 31) ? (u & 0x7FFFFFFFu) : ~u);
}
__device__ inline void sg_edge(const int* es, const int* ed, int e, int& s, int& d) {
  if (e < cE) { s = es[e]; d = ed[e]; } else { s = e - cE; d = s; }
}

__global__ void sg_gat_logit(const int* __restrict__ es, const int* __restrict__ ed,
                             const float* __restrict__ xl, const float* __restrict__ xr,
                             const float* __restrict__ att, float* __restrict__ logit,
                             int h, int c) {
  long long gid = (long long)blockIdx.x * 256 + threadIdx.x;
  long long tot = (long long)cET * h;
  if (gid >= tot) return;
  int e = (int)(gid / h), hh = (int)(gid % h);
  int s, d; sg_edge(es, ed, e, s, d);
  int hc = h * c;
  const float* pl = xl + (long long)s * hc + hh * c;
  const float* pr = xr + (long long)d * hc + hh * c;
  const float* pa = att + hh * c;
  float acc = 0.f;
  for (int i = 0; i < c; ++i) {
    float v = pl[i] + pr[i];
    v = v > 0.f ? v : 0.2f * v;          // leaky_relu(0.2)
    acc += v * pa[i];
  }
  logit[(long long)e * h + hh] = acc;
}

__global__ void sg_gat_segmax(const int* es, const int* ed,
                              const float* __restrict__ logit,
                              unsigned* __restrict__ menc, int h) {
  long long gid = (long long)blockIdx.x * 256 + threadIdx.x;
  if (gid >= (long long)cET * h) return;
  int e = (int)(gid / h), hh = (int)(gid % h);
  int s, d; sg_edge(es, ed, e, s, d);
  atomicMax(&menc[(long long)d * h + hh], sg_flip(logit[gid]));
}

__global__ void sg_gat_exp(const int* es, const int* ed,
                           const float* __restrict__ logit,
                           const unsigned* __restrict__ menc,
                           float* __restrict__ a, float* __restrict__ z, int h) {
  long long gid = (long long)blockIdx.x * 256 + threadIdx.x;
  if (gid >= (long long)cET * h) return;
  int e = (int)(gid / h), hh = (int)(gid % h);
  int s, d; sg_edge(es, ed, e, s, d);
  float m = sg_unflip(menc[(long long)d * h + hh]);
  float v = __expf(logit[gid] - m);
  a[gid] = v;
  atomicAdd(&z[(long long)d * h + hh], v);
}

__global__ void sg_gat_norm(const int* es, const int* ed,
                            float* __restrict__ a, const float* __restrict__ z, int h) {
  long long gid = (long long)blockIdx.x * 256 + threadIdx.x;
  if (gid >= (long long)cET * h) return;
  int e = (int)(gid / h), hh = (int)(gid % h);
  int s, d; sg_edge(es, ed, e, s, d);
  a[gid] /= z[(long long)d * h + hh];
}

__global__ void sg_gat_accum(const int* es, const int* ed,
                             const float* __restrict__ xl, const float* __restrict__ a,
                             float* __restrict__ out, int h, int c) {
  int hc = h * c;
  long long gid = (long long)blockIdx.x * 256 + threadIdx.x;
  long long tot = (long long)cET * hc;
  if (gid >= tot) return;
  int e = (int)(gid / hc), j = (int)(gid % hc);
  int hh = j / c;
  int s, d; sg_edge(es, ed, e, s, d);
  atomicAdd(&out[(long long)d * hc + j],
            xl[(long long)s * hc + j] * a[(long long)e * h + hh]);
}

// BatchNorm over relu(x + bias): stats then apply into a column slice of xc
__global__ void sg_bn_stats(const float* __restrict__ x, const float* __restrict__ bias,
                            float* __restrict__ mean, float* __restrict__ rs, int hc) {
  int ch = blockIdx.x;
  __shared__ float s1[256], s2[256];
  float a = 0.f, b = 0.f;
  for (int n = threadIdx.x; n < cN; n += 256) {
    float v = x[(long long)n * hc + ch] + bias[ch];
    v = v > 0.f ? v : 0.f;
    a += v; b += v * v;
  }
  s1[threadIdx.x] = a; s2[threadIdx.x] = b;
  __syncthreads();
  for (int st = 128; st > 0; st >>= 1) {
    if (threadIdx.x < st) { s1[threadIdx.x] += s1[threadIdx.x + st];
                            s2[threadIdx.x] += s2[threadIdx.x + st]; }
    __syncthreads();
  }
  if (threadIdx.x == 0) {
    float m = s1[0] / cN;
    float var = s2[0] / cN - m * m;
    mean[ch] = m;
    rs[ch] = rsqrtf(var + 1e-5f);
  }
}

__global__ void sg_bn_apply(const float* __restrict__ x, const float* __restrict__ bias,
                            const float* __restrict__ mean, const float* __restrict__ rs,
                            const float* __restrict__ g, const float* __restrict__ be,
                            float* __restrict__ xc, int hc, int coloff) {
  long long gid = (long long)blockIdx.x * 256 + threadIdx.x;
  if (gid >= (long long)cN * hc) return;
  int n = (int)(gid / hc), ch = (int)(gid % hc);
  float v = x[gid] + bias[ch];
  v = v > 0.f ? v : 0.f;
  xc[(long long)n * cD + coloff + ch] = (v - mean[ch]) * rs[ch] * g[ch] + be[ch];
}

// layer-3: mean over 2 heads (+b3) into xc columns 512..544
__global__ void sg_gat3_fin(const float* __restrict__ gout, const float* __restrict__ b3,
                            float* __restrict__ xc) {
  long long gid = (long long)blockIdx.x * 256 + threadIdx.x;
  if (gid >= (long long)cN * 32) return;
  int n = (int)(gid / 32), cc = (int)(gid % 32);
  float v = 0.5f * (gout[(long long)n * 64 + cc] + gout[(long long)n * 64 + 32 + cc]) + b3[cc];
  xc[(long long)n * cD + 512 + cc] = v;
}

__global__ void sg_softmax30(float* __restrict__ s) {
  int n = blockIdx.x * 256 + threadIdx.x;
  if (n >= cN) return;
  float* p = s + (long long)n * cK;
  float m = p[0];
  for (int i = 1; i < cK; ++i) m = fmaxf(m, p[i]);
  float z = 0.f;
  for (int i = 0; i < cK; ++i) { float e = __expf(p[i] - m); p[i] = e; z += e; }
  float inv = 1.f / z;
  for (int i = 0; i < cK; ++i) p[i] *= inv;
}

__global__ void sg_adj_build(const int* __restrict__ es, const int* __restrict__ ed,
                             float* __restrict__ adj) {
  int e = blockIdx.x * 256 + threadIdx.x;
  if (e >= cE) return;
  int s = es[e], d = ed[e];
  int b = s / cNPG;
  atomicAdd(&adj[(long long)b * cNPG * cNPG + (long long)(s % cNPG) * cNPG + (d % cNPG)], 1.0f);
}

__global__ void sg_ap_diag(float* __restrict__ ap) {
  int g = blockIdx.x * 256 + threadIdx.x;
  if (g >= cB * cK) return;
  int b = g / cK, k = g % cK;
  ap[(long long)b * cK * cK + k * cK + k] = 1.0f;
}

__global__ void sg_ap_deg(const float* __restrict__ ap, float* __restrict__ dg) {
  int g = blockIdx.x * 256 + threadIdx.x;
  if (g >= cB * cK) return;
  int b = g / cK, k = g % cK;
  float s = 0.f;
  for (int l = 0; l < cK; ++l) s += ap[(long long)b * cK * cK + k * cK + l];
  dg[g] = rsqrtf(fmaxf(s, 1.0f));
}

// writes APN padded to lda=32 (pad columns pre-zeroed)
__global__ void sg_ap_norm(const float* __restrict__ ap, const float* __restrict__ dg,
                           float* __restrict__ apn) {
  long long g = (long long)blockIdx.x * 256 + threadIdx.x;
  if (g >= (long long)cB * cK * cK) return;
  int b = (int)(g / (cK * cK));
  int r = (int)((g % (cK * cK)) / cK);
  int l = (int)(g % cK);
  apn[(long long)b * cK * 32 + r * 32 + l] = dg[b * cK + r] * dg[b * cK + l] * ap[g];
}

// neighbor-embedding helpers
__global__ void sg_nb_deg(const float* __restrict__ adj, float* __restrict__ dv) {
  int row = blockIdx.x;
  __shared__ float sm[256];
  float a = 0.f;
  for (int j = threadIdx.x; j < cND; j += 256) a += adj[(long long)row * cND + j];
  sm[threadIdx.x] = a;
  __syncthreads();
  for (int st = 128; st > 0; st >>= 1) {
    if (threadIdx.x < st) sm[threadIdx.x] += sm[threadIdx.x + st];
    __syncthreads();
  }
  if (threadIdx.x == 0) dv[row] = rsqrtf(sm[0] + 1.0f);   // +1 for self loop
}

__global__ void sg_nb_scale(const float* __restrict__ h0, const float* __restrict__ dv,
                            float* __restrict__ t) {
  long long g = (long long)blockIdx.x * 256 + threadIdx.x;
  if (g >= (long long)cND * 32) return;
  t[g] = dv[g / 32] * h0[g];
}

__global__ void sg_nb_gather(const float* __restrict__ h0, const float* __restrict__ t,
                             const float* __restrict__ u, const float* __restrict__ dv,
                             const int* __restrict__ idx, float* __restrict__ g, int gofs) {
  int i = blockIdx.x;       // pair index, 0..127
  int c = threadIdx.x;      // channel, 0..31 (one wave)
  int r = idx[i];
  long long o = (long long)r * 32 + c;
  float v = 0.2f * h0[o] + 0.8f * dv[r] * (u[o] + t[o]);
  __shared__ float sm[32];
  sm[c] = v * v;
  __syncthreads();
  for (int st = 16; st > 0; st >>= 1) {
    if (c < st) sm[c] += sm[c + st];
    __syncthreads();
  }
  float nrm = fmaxf(sqrtf(sm[0]), 1e-12f);
  g[(long long)i * 16384 + gofs + c] = v / nrm;
}

__global__ void sg_out(const float* __restrict__ hb, const float* __restrict__ Wl2,
                       const float* __restrict__ bl2, float* __restrict__ out) {
  int i = threadIdx.x;
  if (i >= cB) return;
  float acc = bl2[0];
  for (int j = 0; j < 128; ++j) acc += hb[(long long)i * 128 + j] * Wl2[j];
  out[i] = acc;
}

// ---------------------------------------------------------------------------
// Host-side helpers
// ---------------------------------------------------------------------------
static inline dim3 g1(long long n) { return dim3((unsigned)((n + 255) / 256)); }

template <int FLAGS>
static void gemm_t(hipStream_t st, const float* A, const float* B, const float* bias,
                   float* C, int M, int N, int K, int lda, int ldb, int ldc,
                   long long sA, long long sB, long long sC, int batches) {
  int tiles = ((M + 63) / 64) * ((N + 63) / 64);
  dim3 grid((unsigned)tiles, (unsigned)batches);
  sg_wmma_gemm<FLAGS><<<grid, dim3(128), 0, st>>>(A, B, bias, C, M, N, K,
                                                  lda, ldb, ldc, sA, sB, sC);
}

static void gemm(hipStream_t st, const float* A, const float* B, const float* bias,
                 float* C, int M, int N, int K, int lda, int ldb, int ldc,
                 long long sA, long long sB, long long sC, int batches, int flags) {
  switch (flags) {
    case 0:                  gemm_t<0>(st, A, B, bias, C, M, N, K, lda, ldb, ldc, sA, sB, sC, batches); break;
    case GF_TRANSA:          gemm_t<GF_TRANSA>(st, A, B, bias, C, M, N, K, lda, ldb, ldc, sA, sB, sC, batches); break;
    case GF_BIAS:            gemm_t<GF_BIAS>(st, A, B, bias, C, M, N, K, lda, ldb, ldc, sA, sB, sC, batches); break;
    default:                 gemm_t<GF_BIAS | GF_RELU>(st, A, B, bias, C, M, N, K, lda, ldb, ldc, sA, sB, sC, batches); break;
  }
}

static void gat_layer(hipStream_t st, const int* es, const int* ed,
                      const float* xl, const float* xr, const float* att,
                      float* logit, float* anrm, float* menc, float* z, float* gout,
                      int h, int c) {
  int hc = h * c;
  long long ETh = (long long)cET * h;
  sg_zero<<<g1((long long)cN * h), 256, 0, st>>>(menc, (long long)cN * h);  // 0u == flip(-huge)
  sg_zero<<<g1((long long)cN * h), 256, 0, st>>>(z, (long long)cN * h);
  sg_zero<<<g1((long long)cN * hc), 256, 0, st>>>(gout, (long long)cN * hc);
  sg_gat_logit<<<g1(ETh), 256, 0, st>>>(es, ed, xl, xr, att, logit, h, c);
  sg_gat_segmax<<<g1(ETh), 256, 0, st>>>(es, ed, logit, (unsigned*)menc, h);
  sg_gat_exp<<<g1(ETh), 256, 0, st>>>(es, ed, logit, (const unsigned*)menc, anrm, z, h);
  sg_gat_norm<<<g1(ETh), 256, 0, st>>>(es, ed, anrm, z, h);
  sg_gat_accum<<<g1((long long)cET * hc), 256, 0, st>>>(es, ed, xl, anrm, gout, h, c);
}

// Workspace layout (floats). Pool-phase buffers reuse the XL/XR/GOUT region.
namespace {
constexpr long long SZ_XL   = (long long)cN * 384;
constexpr long long OFF_XL  = 0;
constexpr long long OFF_XR  = OFF_XL + SZ_XL;
constexpr long long OFF_GO  = OFF_XR + SZ_XL;
constexpr long long OFF_XC  = OFF_GO + SZ_XL;
constexpr long long OFF_LG  = OFF_XC + (long long)cN * cD;
constexpr long long OFF_AN  = OFF_LG + (long long)cET * 3;
constexpr long long OFF_ME  = OFF_AN + (long long)cET * 3;
constexpr long long OFF_Z   = OFF_ME + (long long)cN * 3;
constexpr long long OFF_S   = OFF_Z + (long long)cN * 3;
constexpr long long OFF_BN  = OFF_S + (long long)cN * cK;     // 768 floats (mean|rs)
// Pool-phase reuse (inside [0, OFF_XC))
constexpr long long P_ADJ = 0;
constexpr long long P_XP  = P_ADJ + (long long)cB * cNPG * cNPG;
constexpr long long P_T2  = P_XP + (long long)cB * cK * cD;
constexpr long long P_AP  = P_T2 + (long long)cB * cK * cNPG;
constexpr long long P_APN = P_AP + (long long)cB * cK * cK;      // padded: B x 30 x 32
constexpr long long P_DG  = P_APN + (long long)cB * cK * 32;
constexpr long long P_XPW = P_DG + (long long)cB * cK;
constexpr long long P_G   = P_XPW + (long long)cB * cK * cD;
constexpr long long P_H0  = P_G + (long long)cB * 16384;
constexpr long long P_T   = P_H0 + (long long)cND * 32;
constexpr long long P_U   = P_T + (long long)cND * 32;
constexpr long long P_DV  = P_U + (long long)cND * 32;
constexpr long long P_HB  = P_DV + cND;
}

extern "C" void kernel_launch(void* const* d_in, const int* in_sizes, int n_in,
                              void* d_out, int out_size, void* d_ws, size_t ws_size,
                              hipStream_t stream) {
  (void)in_sizes; (void)n_in; (void)out_size; (void)ws_size;
  const float* x        = (const float*)d_in[0];
  const int*   esrc     = (const int*)d_in[1];
  const int*   edst     = (const int*)d_in[2];
  const int*   pairs1   = (const int*)d_in[4];
  const int*   pairs2   = (const int*)d_in[5];
  const float* drug_adj = (const float*)d_in[6];
  const float* drug_emb = (const float*)d_in[7];
  const float* dis_adj  = (const float*)d_in[8];
  const float* dis_emb  = (const float*)d_in[9];
  const float* Wde = (const float*)d_in[10]; const float* bde = (const float*)d_in[11];
  const float* Wse = (const float*)d_in[12]; const float* bse = (const float*)d_in[13];
  const float* W1l = (const float*)d_in[14]; const float* W1r = (const float*)d_in[15];
  const float* att1= (const float*)d_in[16]; const float* b1  = (const float*)d_in[17];
  const float* W2l = (const float*)d_in[18]; const float* W2r = (const float*)d_in[19];
  const float* att2= (const float*)d_in[20]; const float* b2  = (const float*)d_in[21];
  const float* W3l = (const float*)d_in[22]; const float* W3r = (const float*)d_in[23];
  const float* att3= (const float*)d_in[24]; const float* b3  = (const float*)d_in[25];
  const float* gm1 = (const float*)d_in[26]; const float* be1 = (const float*)d_in[27];
  const float* gm2 = (const float*)d_in[28]; const float* be2 = (const float*)d_in[29];
  const float* Ws  = (const float*)d_in[30]; const float* bs  = (const float*)d_in[31];
  const float* Wg  = (const float*)d_in[32]; const float* bg  = (const float*)d_in[33];
  const float* Wl1 = (const float*)d_in[34]; const float* bl1 = (const float*)d_in[35];
  const float* Wl2 = (const float*)d_in[36]; const float* bl2 = (const float*)d_in[37];
  float* out = (float*)d_out;
  float* W   = (float*)d_ws;

  float* XL = W + OFF_XL;  float* XR = W + OFF_XR;  float* GO = W + OFF_GO;
  float* XC = W + OFF_XC;  float* LG = W + OFF_LG;  float* AN = W + OFF_AN;
  float* ME = W + OFF_ME;  float* Z  = W + OFF_Z;   float* S  = W + OFF_S;
  float* MEAN = W + OFF_BN; float* RS = W + OFF_BN + 384;

  // ---- GAT layer 1: hc=384 (h=3,c=128) ----
  gemm(stream, x, W1l, nullptr, XL, cN, 384, cF, cF, 384, 384, 0, 0, 0, 1, 0);
  gemm(stream, x, W1r, nullptr, XR, cN, 384, cF, cF, 384, 384, 0, 0, 0, 1, 0);
  gat_layer(stream, esrc, edst, XL, XR, att1, LG, AN, ME, Z, GO, 3, 128);
  sg_bn_stats<<<dim3(384), dim3(256), 0, stream>>>(GO, b1, MEAN, RS, 384);
  sg_bn_apply<<<g1((long long)cN * 384), 256, 0, stream>>>(GO, b1, MEAN, RS, gm1, be1, XC, 384, 0);

  // ---- GAT layer 2: in = xc[:, :384], hc=128 (h=2,c=64) ----
  gemm(stream, XC, W2l, nullptr, XL, cN, 128, 384, cD, 128, 128, 0, 0, 0, 1, 0);
  gemm(stream, XC, W2r, nullptr, XR, cN, 128, 384, cD, 128, 128, 0, 0, 0, 1, 0);
  gat_layer(stream, esrc, edst, XL, XR, att2, LG, AN, ME, Z, GO, 2, 64);
  sg_bn_stats<<<dim3(128), dim3(256), 0, stream>>>(GO, b2, MEAN, RS, 128);
  sg_bn_apply<<<g1((long long)cN * 128), 256, 0, stream>>>(GO, b2, MEAN, RS, gm2, be2, XC, 128, 384);

  // ---- GAT layer 3: in = xc[:, 384:512], hc=64 (h=2,c=32), mean heads ----
  gemm(stream, XC + 384, W3l, nullptr, XL, cN, 64, 128, cD, 64, 64, 0, 0, 0, 1, 0);
  gemm(stream, XC + 384, W3r, nullptr, XR, cN, 64, 128, cD, 64, 64, 0, 0, 0, 1, 0);
  gat_layer(stream, esrc, edst, XL, XR, att3, LG, AN, ME, Z, GO, 2, 32);
  sg_gat3_fin<<<g1((long long)cN * 32), 256, 0, stream>>>(GO, b3, XC);

  // ---- pooling assignment s = softmax(xc @ Ws + bs) ----
  gemm(stream, XC, Ws, bs, S, cN, cK, cD, cD, cK, cK, 0, 0, 0, 1, GF_BIAS);
  sg_softmax30<<<g1(cN), 256, 0, stream>>>(S);

  // ---- dense per-graph adjacency + pooled graph ----
  float* ADJ = W + P_ADJ;  float* XP = W + P_XP;   float* T2 = W + P_T2;
  float* AP  = W + P_AP;   float* APN = W + P_APN; float* DG = W + P_DG;
  float* XPW = W + P_XPW;  float* G  = W + P_G;
  float* H0  = W + P_H0;   float* Tb = W + P_T;    float* U = W + P_U;
  float* DV  = W + P_DV;   float* HB = W + P_HB;

  sg_zero<<<g1((long long)cB * cNPG * cNPG), 256, 0, stream>>>(ADJ, (long long)cB * cNPG * cNPG);
  sg_adj_build<<<g1(cE), 256, 0, stream>>>(esrc, edst, ADJ);

  // xp[b] = s_b^T @ xd_b      (30x128 @ 128x544)
  gemm(stream, S, XC, nullptr, XP, cK, cD, cNPG, cK, cD, cD,
       (long long)cNPG * cK, (long long)cNPG * cD, (long long)cK * cD, cB, GF_TRANSA);
  // t2[b] = s_b^T @ adj_b     (30x128 @ 128x128)
  gemm(stream, S, ADJ, nullptr, T2, cK, cNPG, cNPG, cK, cNPG, cNPG,
       (long long)cNPG * cK, (long long)cNPG * cNPG, (long long)cK * cNPG, cB, GF_TRANSA);
  // ap[b] = t2_b @ s_b        (30x128 @ 128x30)
  gemm(stream, T2, S, nullptr, AP, cK, cK, cNPG, cNPG, cK, cK,
       (long long)cK * cNPG, (long long)cNPG * cK, (long long)cK * cK, cB, 0);
  sg_ap_diag<<<g1(cB * cK), 256, 0, stream>>>(AP);
  sg_ap_deg<<<g1(cB * cK), 256, 0, stream>>>(AP, DG);
  sg_zero<<<g1((long long)cB * cK * 32), 256, 0, stream>>>(APN, (long long)cB * cK * 32);
  sg_ap_norm<<<g1((long long)cB * cK * cK), 256, 0, stream>>>(AP, DG, APN);
  // xpw[b] = xp_b @ Wg        (30x544 @ 544x544)
  gemm(stream, XP, Wg, nullptr, XPW, cK, cD, cD, cD, cD, cD,
       (long long)cK * cD, 0, (long long)cK * cD, cB, 0);
  // xg[b] = relu(apn_b @ xpw_b + bg) -> written straight into g[:, 32:16352]
  // (APN K-padded to 32 with zeros; pad rows of XPW multiply by exactly 0)
  gemm(stream, APN, XPW, bg, G + 32, cK, cD, 32, 32, cD, cD,
       (long long)cK * 32, (long long)cK * cD, 16384LL, cB, GF_BIAS | GF_RELU);

  // ---- neighbor embeddings (drug -> g[:,0:32], disease -> g[:,16352:16384]) ----
  for (int which = 0; which < 2; ++which) {
    const float* emb = which ? dis_emb : drug_emb;
    const float* adj = which ? dis_adj : drug_adj;
    const float* Wn  = which ? Wse : Wde;
    const float* bn  = which ? bse : bde;
    const int* idx   = which ? pairs2 : pairs1;
    int gofs         = which ? (32 + cK * cD) : 0;   // 16352 or 0
    gemm(stream, emb, Wn, bn, H0, cND, 32, cND, cND, 32, 32, 0, 0, 0, 1, GF_BIAS);
    sg_nb_deg<<<dim3(cND), dim3(256), 0, stream>>>(adj, DV);
    sg_nb_scale<<<g1((long long)cND * 32), 256, 0, stream>>>(H0, DV, Tb);
    gemm(stream, adj, Tb, nullptr, U, cND, 32, cND, cND, 32, 32, 0, 0, 0, 1, 0);
    sg_nb_gather<<<dim3(cB), dim3(32), 0, stream>>>(H0, Tb, U, DV, idx, G, gofs);
  }

  // ---- final MLP ----
  gemm(stream, G, Wl1, bl1, HB, cB, 128, 16384, 16384, 128, 128, 0, 0, 0, 1,
       GF_BIAS | GF_RELU);
  sg_out<<<dim3(1), dim3(128), 0, stream>>>(HB, Wl2, bl2, out);
}
